// MXMLocalMP_24953759989849
// MI455X (gfx1250) — matrix operand order repaired and checked by reading, not compiled, not run on hardware
//
#include <hip/hip_runtime.h>
#include <hip/hip_bf16.h>

// ---------------------------------------------------------------------------
// MXMLocalMP (DimeNet-style block) for gfx1250 / MI455X.
// bf16 WMMA (f32 accumulate) for every GEMM; fp32 HW atomics for segment sums.
// Weights staged in LDS pre-swizzled into B-fragment order (one 32B chunk per
// lane per fragment -> 2x ds_load_b128). High-pressure edge kernels keep the
// column-tile loop rolled to avoid scratch spills.
// ---------------------------------------------------------------------------

#define NODES  30000
#define EDGES  480000
#define ANGLES 960000

typedef __attribute__((ext_vector_type(16))) __bf16 v16bf;
typedef __attribute__((ext_vector_type(8)))  __bf16 v8bf;
typedef __attribute__((ext_vector_type(8)))  float  v8f;

__device__ __forceinline__ __bf16 f2bf(float x) { return (__bf16)x; }

// silu via hardware rcp/exp (v_exp_f32 + v_rcp_f32), no IEEE divide expansion
__device__ __forceinline__ float silu_f(float x) {
    const float e = __expf(-x);
    return x * __builtin_amdgcn_rcpf(1.0f + e);
}

__device__ __forceinline__ v8f zero8() {
    v8f z;
#pragma unroll
    for (int i = 0; i < 8; ++i) z[i] = 0.0f;
    return z;
}

__device__ __forceinline__ v8f wmma_bf16(v16bf a, v16bf b, v8f c) {
    return __builtin_amdgcn_wmma_f32_16x16x32_bf16(
        /*neg_a=*/false, a, /*neg_b=*/false, b,
        /*c_mod=*/(short)0, c, /*reuse_a=*/false, /*reuse_b=*/false);
}

// ---------------------------------------------------------------------------
// A-fragment (16x32 bf16). Caller passes p = rowBase + 8*laneHi (+ kOffset).
// CDNA5 layout: per lane the 16 halves are K = {0..7} and {16..23} (lanes
// 16-31 shifted by +8) -> two contiguous runs -> vector loads.
// ---------------------------------------------------------------------------
__device__ __forceinline__ v16bf load_a_core_f32(const float* p) {
    const float4 x0 = *(const float4*)(p);
    const float4 x1 = *(const float4*)(p + 4);
    const float4 x2 = *(const float4*)(p + 16);
    const float4 x3 = *(const float4*)(p + 20);
    v16bf a;
    a[0]  = f2bf(x0.x); a[1]  = f2bf(x0.y); a[2]  = f2bf(x0.z); a[3]  = f2bf(x0.w);
    a[4]  = f2bf(x1.x); a[5]  = f2bf(x1.y); a[6]  = f2bf(x1.z); a[7]  = f2bf(x1.w);
    a[8]  = f2bf(x2.x); a[9]  = f2bf(x2.y); a[10] = f2bf(x2.z); a[11] = f2bf(x2.w);
    a[12] = f2bf(x3.x); a[13] = f2bf(x3.y); a[14] = f2bf(x3.z); a[15] = f2bf(x3.w);
    return a;
}

__device__ __forceinline__ v16bf load_a_core_bf16(const __bf16* p) {
    const v8bf lo = *(const v8bf*)(p);
    const v8bf hi = *(const v8bf*)(p + 16);
    return __builtin_shufflevector(lo, hi, 0, 1, 2, 3, 4, 5, 6, 7,
                                   8, 9, 10, 11, 12, 13, 14, 15);
}

// ---------------------------------------------------------------------------
// Swizzled weight staging: source W is (krows x 64) f32 row-major.
// Fragment f = kb*4 + ct holds the (32 x 16) B tile; lane (hi*16+col) owns
// 16 contiguous halves (K = kb*32 + 8*hi + {0..7, 16..23} at column ct*16+col).
// ---------------------------------------------------------------------------
__device__ __forceinline__ void stage_w_sw(__bf16* dst, const float* W,
                                           int krows, int tid, int nthr) {
    const int total = krows * 64;
    for (int idx = tid; idx < total; idx += nthr) {
        const int k = idx >> 6;
        const int n = idx & 63;
        const int kb = k >> 5;
        const int hi = (k >> 3) & 1;
        const int i  = (k & 7) | ((k >> 1) & 8);
        const int ct = n >> 4;
        const int col = n & 15;
        dst[(((kb * 4 + ct) * 32) + (hi * 16 + col)) * 16 + i] = f2bf(W[idx]);
    }
}

// One 32-byte contiguous LDS read per lane -> 2x ds_load_b128
__device__ __forceinline__ v16bf load_b_sw(const __bf16* sw, int kb, int ct) {
    const int lane = threadIdx.x & 31;
    return *(const v16bf*)(sw + (((kb * 4 + ct) * 32) + lane) * 16);
}

__device__ __forceinline__ void atomAddF(float* p, float v) {
    unsafeAtomicAdd(p, v);
}

// ---------------------------------------------------------------------------
// Generic node-level GEMM: out = silu(A @ W + bias) [+ resid].  rows x 64.
// 128 threads = 4 waves, each wave owns one 16-row tile.
// ---------------------------------------------------------------------------
template <bool HAS_RESID>
__global__ void k_node_gemm(const float* __restrict__ A,
                            const float* __restrict__ W,
                            const float* __restrict__ bias,
                            const float* __restrict__ resid,
                            float* __restrict__ out, int rows) {
    __shared__ alignas(32) __bf16 sW[8 * 32 * 16];
    const int tid = threadIdx.x;
    stage_w_sw(sW, W, 64, tid, blockDim.x);
    __syncthreads();

    const int w = tid >> 5, lane = tid & 31;
    const int t = blockIdx.x * 4 + w;
    if (t * 16 >= rows) return;
    const int r0 = t * 16;
    const int hi = lane >> 4, rsel = lane & 15;

    const float* rp = A + (size_t)(r0 + rsel) * 64 + 8 * hi;
    const v16bf a0 = load_a_core_f32(rp);
    const v16bf a1 = load_a_core_f32(rp + 32);

#pragma unroll
    for (int ct = 0; ct < 4; ++ct) {
        v8f acc = zero8();
        acc = wmma_bf16(a0, load_b_sw(sW, 0, ct), acc);
        acc = wmma_bf16(a1, load_b_sw(sW, 1, ct), acc);
        const int col = ct * 16 + rsel;
        const float bv = bias[col];
#pragma unroll
        for (int r = 0; r < 8; ++r) {
            const int m = r + 8 * hi;
            const size_t o = (size_t)(r0 + m) * 64 + col;
            float v = silu_f(acc[r] + bv);
            if (HAS_RESID) v += resid[o];
            out[o] = v;
        }
    }
}

// ---------------------------------------------------------------------------
// Edge stage 1: m = [h1[ei0] | h1[ei1] | rbf]  (M x 192)
//   mkj = silu(m @ kj_W + kj_b) * (rbf @ rbf1_W)
//   m1  = silu(m @ ji1_W + ji1_b)          (seed for angle-1 scatter-add)
// ct loop kept rolled: 6 live A frags + 3 accumulators must not be
// interleaved across column tiles or the kernel spills to scratch.
// ---------------------------------------------------------------------------
__global__ void k_edge1(const float* __restrict__ h1,
                        const float* __restrict__ rbf,
                        const int*   __restrict__ ei,
                        const float* __restrict__ kjW, const float* __restrict__ kjb,
                        const float* __restrict__ jiW, const float* __restrict__ jib,
                        const float* __restrict__ r1W,
                        float* __restrict__ mkj, float* __restrict__ m1) {
    __shared__ alignas(32) __bf16 sKJ[24 * 32 * 16];
    __shared__ alignas(32) __bf16 sJI[24 * 32 * 16];
    __shared__ alignas(32) __bf16 sR1[8 * 32 * 16];
    const int tid = threadIdx.x;
    stage_w_sw(sKJ, kjW, 192, tid, blockDim.x);
    stage_w_sw(sJI, jiW, 192, tid, blockDim.x);
    stage_w_sw(sR1, r1W, 64, tid, blockDim.x);
    __syncthreads();

    const int w = tid >> 5, lane = tid & 31;
    const int t = blockIdx.x * 4 + w;
    const int e0 = t * 16;
    const int hi = lane >> 4, rsel = lane & 15;
    const int e = e0 + rsel;

    const int* ei0 = ei;
    const int* ei1 = ei + EDGES;
    const int ri = ei0[e], rj = ei1[e];

    v16bf A[6];
    {
        const float* p = h1 + (size_t)ri * 64 + 8 * hi;
        A[0] = load_a_core_f32(p);  A[1] = load_a_core_f32(p + 32);
    }
    {
        const float* p = h1 + (size_t)rj * 64 + 8 * hi;
        A[2] = load_a_core_f32(p);  A[3] = load_a_core_f32(p + 32);
    }
    {
        const float* p = rbf + (size_t)e * 64 + 8 * hi;
        A[4] = load_a_core_f32(p);  A[5] = load_a_core_f32(p + 32);
    }

#pragma unroll 1
    for (int ct = 0; ct < 4; ++ct) {
        v8f aKJ = zero8(), aJI = zero8(), aR = zero8();
#pragma unroll
        for (int kb = 0; kb < 6; ++kb) {
            aKJ = wmma_bf16(A[kb], load_b_sw(sKJ, kb, ct), aKJ);
            aJI = wmma_bf16(A[kb], load_b_sw(sJI, kb, ct), aJI);
        }
        aR = wmma_bf16(A[4], load_b_sw(sR1, 0, ct), aR);
        aR = wmma_bf16(A[5], load_b_sw(sR1, 1, ct), aR);

        const int col = ct * 16 + rsel;
        const float bkj = kjb[col], bji = jib[col];
#pragma unroll
        for (int r = 0; r < 8; ++r) {
            const int m = r + 8 * hi;
            const size_t o = (size_t)(e0 + m) * 64 + col;
            mkj[o] = silu_f(aKJ[r] + bkj) * aR[r];
            m1[o]  = silu_f(aJI[r] + bji);
        }
    }
}

// ---------------------------------------------------------------------------
// Angle stage: sw = silu(silu(sbf@W1+b1)@W2+b2); dst[gid0] += sw*src[gid1]
// Two chained WMMA GEMMs; intermediate routed through per-wave LDS as bf16.
// ---------------------------------------------------------------------------
__global__ void k_angle(const float* __restrict__ sbf,
                        const float* __restrict__ W1, const float* __restrict__ b1,
                        const float* __restrict__ W2, const float* __restrict__ b2,
                        const float* __restrict__ src, float* __restrict__ dst,
                        const int* __restrict__ gidxArr,   // angle_idx[1] (gather)
                        const int* __restrict__ didxArr) { // angle_idx[0] (scatter)
    __shared__ alignas(32) __bf16 sW1[8 * 32 * 16];
    __shared__ alignas(32) __bf16 sW2[8 * 32 * 16];
    __shared__ alignas(32) __bf16 sInter[4][16 * 64];
    const int tid = threadIdx.x;
    stage_w_sw(sW1, W1, 64, tid, blockDim.x);
    stage_w_sw(sW2, W2, 64, tid, blockDim.x);
    __syncthreads();

    const int w = tid >> 5, lane = tid & 31;
    const int t = blockIdx.x * 4 + w;
    const int a0 = t * 16;
    const int hi = lane >> 4, rsel = lane & 15;

    // GEMM 1: silu(sbf @ W1 + b1) -> LDS (bf16, row-major 16x64, wave-private)
    {
        const float* p = sbf + (size_t)(a0 + rsel) * 64 + 8 * hi;
        const v16bf A0 = load_a_core_f32(p);
        const v16bf A1 = load_a_core_f32(p + 32);
        __bf16* inter = &sInter[w][0];
#pragma unroll
        for (int ct = 0; ct < 4; ++ct) {
            v8f acc = zero8();
            acc = wmma_bf16(A0, load_b_sw(sW1, 0, ct), acc);
            acc = wmma_bf16(A1, load_b_sw(sW1, 1, ct), acc);
            const int col = ct * 16 + rsel;
            const float bv = b1[col];
#pragma unroll
            for (int r = 0; r < 8; ++r) {
                const int m = r + 8 * hi;
                inter[m * 64 + col] = f2bf(silu_f(acc[r] + bv));
            }
        }
    }
    __syncthreads();

    // GEMM 2 + fused gather * multiply * atomic scatter-add
    const __bf16* ip = &sInter[w][0] + rsel * 64 + 8 * hi;
    const v16bf B0 = load_a_core_bf16(ip);
    const v16bf B1 = load_a_core_bf16(ip + 32);

    int gidx[8], didx[8];
#pragma unroll
    for (int r = 0; r < 8; ++r) {
        const int m = a0 + r + 8 * hi;
        gidx[r] = gidxArr[m];
        didx[r] = didxArr[m];
    }

#pragma unroll
    for (int ct = 0; ct < 4; ++ct) {
        v8f acc = zero8();
        acc = wmma_bf16(B0, load_b_sw(sW2, 0, ct), acc);
        acc = wmma_bf16(B1, load_b_sw(sW2, 1, ct), acc);
        const int col = ct * 16 + rsel;
        const float bv = b2[col];
#pragma unroll
        for (int r = 0; r < 8; ++r) {
            const float sw  = silu_f(acc[r] + bv);
            const float val = sw * src[(size_t)gidx[r] * 64 + col];
            atomAddF(&dst[(size_t)didx[r] * 64 + col], val);
        }
    }
}

// ---------------------------------------------------------------------------
// Edge stage 2 (m1 may alias m2out — all loads precede stores via dataflow):
//   mjj = silu(m1 @ jj_W + jj_b) * (rbf @ rbf2_W)
//   m2  = silu(m1 @ ji2_W + ji2_b)
// ct loop rolled (4 live A frags + 3 accumulators).
// ---------------------------------------------------------------------------
__global__ void k_edge2(const float* m1, const float* __restrict__ rbf,
                        const float* __restrict__ jjW,  const float* __restrict__ jjb,
                        const float* __restrict__ ji2W, const float* __restrict__ ji2b,
                        const float* __restrict__ r2W,
                        float* mjj, float* m2) {
    __shared__ alignas(32) __bf16 sJJ[8 * 32 * 16];
    __shared__ alignas(32) __bf16 sI2[8 * 32 * 16];
    __shared__ alignas(32) __bf16 sR2[8 * 32 * 16];
    const int tid = threadIdx.x;
    stage_w_sw(sJJ, jjW, 64, tid, blockDim.x);
    stage_w_sw(sI2, ji2W, 64, tid, blockDim.x);
    stage_w_sw(sR2, r2W, 64, tid, blockDim.x);
    __syncthreads();

    const int w = tid >> 5, lane = tid & 31;
    const int t = blockIdx.x * 4 + w;
    const int e0 = t * 16;
    const int hi = lane >> 4, rsel = lane & 15;

    const float* pm = m1 + (size_t)(e0 + rsel) * 64 + 8 * hi;
    const v16bf M0 = load_a_core_f32(pm);
    const v16bf M1 = load_a_core_f32(pm + 32);
    const float* pr = rbf + (size_t)(e0 + rsel) * 64 + 8 * hi;
    const v16bf R0 = load_a_core_f32(pr);
    const v16bf R1 = load_a_core_f32(pr + 32);

#pragma unroll 1
    for (int ct = 0; ct < 4; ++ct) {
        v8f aJ = zero8(), aI = zero8(), aR = zero8();
        aJ = wmma_bf16(M0, load_b_sw(sJJ, 0, ct), aJ);
        aJ = wmma_bf16(M1, load_b_sw(sJJ, 1, ct), aJ);
        aI = wmma_bf16(M0, load_b_sw(sI2, 0, ct), aI);
        aI = wmma_bf16(M1, load_b_sw(sI2, 1, ct), aI);
        aR = wmma_bf16(R0, load_b_sw(sR2, 0, ct), aR);
        aR = wmma_bf16(R1, load_b_sw(sR2, 1, ct), aR);

        const int col = ct * 16 + rsel;
        const float bj = jjb[col], bi = ji2b[col];
#pragma unroll
        for (int r = 0; r < 8; ++r) {
            const int m = r + 8 * hi;
            const size_t o = (size_t)(e0 + m) * 64 + col;
            mjj[o] = silu_f(aJ[r] + bj) * aR[r];
            m2[o]  = silu_f(aI[r] + bi);
        }
    }
}

// ---------------------------------------------------------------------------
// Edge -> node aggregation:  hn[ei0[e]] += (rbf @ rbfo_W)[e] * m2[e]
// ---------------------------------------------------------------------------
__global__ void k_edge_out(const float* __restrict__ rbf,
                           const float* __restrict__ roW,
                           const float* __restrict__ m2,
                           const int*   __restrict__ ei0,
                           float* __restrict__ hn) {
    __shared__ alignas(32) __bf16 sRO[8 * 32 * 16];
    const int tid = threadIdx.x;
    stage_w_sw(sRO, roW, 64, tid, blockDim.x);
    __syncthreads();

    const int w = tid >> 5, lane = tid & 31;
    const int t = blockIdx.x * 4 + w;
    const int e0 = t * 16;
    const int hi = lane >> 4, rsel = lane & 15;

    const float* pr = rbf + (size_t)(e0 + rsel) * 64 + 8 * hi;
    const v16bf R0 = load_a_core_f32(pr);
    const v16bf R1 = load_a_core_f32(pr + 32);

    int nodes[8];
#pragma unroll
    for (int r = 0; r < 8; ++r) nodes[r] = ei0[e0 + r + 8 * hi];

#pragma unroll
    for (int ct = 0; ct < 4; ++ct) {
        v8f acc = zero8();
        acc = wmma_bf16(R0, load_b_sw(sRO, 0, ct), acc);
        acc = wmma_bf16(R1, load_b_sw(sRO, 1, ct), acc);
        const int col = ct * 16 + rsel;
#pragma unroll
        for (int r = 0; r < 8; ++r) {
            const int m = r + 8 * hi;
            const float val = acc[r] * m2[(size_t)(e0 + m) * 64 + col];
            atomAddF(&hn[(size_t)nodes[r] * 64 + col], val);
        }
    }
}

// Final projection: y[i] = y3[i,:] @ yo_W + yo_b   (64 -> 1)
__global__ void k_ydot(const float* __restrict__ y3,
                       const float* __restrict__ yoW,
                       const float* __restrict__ yob,
                       float* __restrict__ y) {
    const int i = blockIdx.x * blockDim.x + threadIdx.x;
    if (i >= NODES) return;
    float acc = yob[0];
    const float* row = y3 + (size_t)i * 64;
#pragma unroll
    for (int k = 0; k < 64; ++k) acc += row[k] * yoW[k];
    y[i] = acc;
}

// ---------------------------------------------------------------------------
extern "C" void kernel_launch(void* const* d_in, const int* in_sizes, int n_in,
                              void* d_out, int out_size, void* d_ws, size_t ws_size,
                              hipStream_t stream) {
    // Inputs (setup_inputs dict order, params flattened in insertion order)
    const float* h_in  = (const float*)d_in[0];
    const float* rbf   = (const float*)d_in[1];
    const float* sbf1  = (const float*)d_in[2];
    const float* sbf2  = (const float*)d_in[3];
    const int*   eidx  = (const int*)d_in[4];
    const int*   aidx1 = (const int*)d_in[5];
    const int*   aidx2 = (const int*)d_in[6];
    const float* h_W   = (const float*)d_in[7];
    const float* h_b   = (const float*)d_in[8];
    const float* kj_W  = (const float*)d_in[9];
    const float* kj_b  = (const float*)d_in[10];
    const float* ji1_W = (const float*)d_in[11];
    const float* ji1_b = (const float*)d_in[12];
    const float* jj_W  = (const float*)d_in[13];
    const float* jj_b  = (const float*)d_in[14];
    const float* ji2_W = (const float*)d_in[15];
    const float* ji2_b = (const float*)d_in[16];
    const float* s1W1  = (const float*)d_in[17];
    const float* s1b1  = (const float*)d_in[18];
    const float* s1W2  = (const float*)d_in[19];
    const float* s1b2  = (const float*)d_in[20];
    const float* s2W1  = (const float*)d_in[21];
    const float* s2b1  = (const float*)d_in[22];
    const float* s2W2  = (const float*)d_in[23];
    const float* s2b2  = (const float*)d_in[24];
    const float* rbf1W = (const float*)d_in[25];
    const float* rbf2W = (const float*)d_in[26];
    const float* rbfoW = (const float*)d_in[27];
    const float* r1W1  = (const float*)d_in[28];
    const float* r1b1  = (const float*)d_in[29];
    const float* r1W2  = (const float*)d_in[30];
    const float* r1b2  = (const float*)d_in[31];
    const float* r2W1  = (const float*)d_in[32];
    const float* r2b1  = (const float*)d_in[33];
    const float* r2W2  = (const float*)d_in[34];
    const float* r2b2  = (const float*)d_in[35];
    const float* r3W1  = (const float*)d_in[36];
    const float* r3b1  = (const float*)d_in[37];
    const float* r3W2  = (const float*)d_in[38];
    const float* r3b2  = (const float*)d_in[39];
    const float* yW1   = (const float*)d_in[40];
    const float* yb1   = (const float*)d_in[41];
    const float* yW2   = (const float*)d_in[42];
    const float* yb2   = (const float*)d_in[43];
    const float* yW3   = (const float*)d_in[44];
    const float* yb3   = (const float*)d_in[45];
    const float* yoW   = (const float*)d_in[46];
    const float* yob   = (const float*)d_in[47];

    // Workspace layout (floats)
    float* ws = (float*)d_ws;
    const size_t nN = (size_t)NODES * 64;
    const size_t nM = (size_t)EDGES * 64;
    float* h1  = ws;                 // (N,64) post h-MLP
    float* mkj = h1 + nN;            // (M,64) edge msg branch-1 ; reused as mjj
    float* m1  = mkj + nM;           // (M,64) m (seeded with m_ji_1) ; reused as m2
    float* hn  = m1 + nM;            // (N,64) node aggregate
    float* nA  = hn + nN;            // (N,64) scratch
    float* nB  = nA + nN;            // (N,64) scratch

    float* out_h = (float*)d_out;            // (N,64)
    float* out_y = out_h + nN;               // (N,)

    const dim3 blk(128);
    const dim3 gN((NODES / 16 + 3) / 4);     // 469 blocks, tail guarded
    const dim3 gE(EDGES / 16 / 4);           // 7500
    const dim3 gA(ANGLES / 16 / 4);          // 15000

    // 1. h1 = silu(h @ h_W + h_b)
    k_node_gemm<false><<<gN, blk, 0, stream>>>(h_in, h_W, h_b, nullptr, h1, NODES);

    // 2. edge branch-1 GEMMs (mkj pre-gather; m1 seeded with m_ji_1)
    k_edge1<<<gE, blk, 0, stream>>>(h1, rbf, eidx, kj_W, kj_b, ji1_W, ji1_b,
                                    rbf1W, mkj, m1);

    // 3. angle-1: m1 += segment_sum(sw1 * mkj[aidx1[1]], aidx1[0])
    k_angle<<<gA, blk, 0, stream>>>(sbf1, s1W1, s1b1, s1W2, s1b2,
                                    mkj, m1, aidx1 + ANGLES, aidx1);

    // 4. edge branch-2 GEMMs (mjj into mkj's buffer; m2 aliases m1's buffer)
    k_edge2<<<gE, blk, 0, stream>>>(m1, rbf, jj_W, jj_b, ji2_W, ji2_b,
                                    rbf2W, mkj, m1);

    // 5. angle-2: m2 += segment_sum(sw2 * mjj[aidx2[1]], aidx2[0])
    k_angle<<<gA, blk, 0, stream>>>(sbf2, s2W1, s2b1, s2W2, s2b2,
                                    mkj, m1, aidx2 + ANGLES, aidx2);

    // 6. hn = segment_sum((rbf @ rbfo_W) * m2, ei0)
    hipMemsetAsync(hn, 0, nN * sizeof(float), stream);
    k_edge_out<<<gE, blk, 0, stream>>>(rbf, rbfoW, m1, eidx, hn);

    // 7-13. node-level residual MLP chain
    k_node_gemm<false><<<gN, blk, 0, stream>>>(hn, r1W1, r1b1, nullptr, nA, NODES);
    k_node_gemm<true><<<gN, blk, 0, stream>>>(nA, r1W2, r1b2, hn,      nB, NODES);
    k_node_gemm<true><<<gN, blk, 0, stream>>>(nB, h_W,  h_b,  h_in,    nA, NODES);
    k_node_gemm<false><<<gN, blk, 0, stream>>>(nA, r2W1, r2b1, nullptr, hn, NODES);
    k_node_gemm<true><<<gN, blk, 0, stream>>>(hn, r2W2, r2b2, nA,      nB, NODES);
    k_node_gemm<false><<<gN, blk, 0, stream>>>(nB, r3W1, r3b1, nullptr, hn, NODES);
    k_node_gemm<true><<<gN, blk, 0, stream>>>(hn, r3W2, r3b2, nB,   out_h, NODES);

    // 14-17. output head
    k_node_gemm<false><<<gN, blk, 0, stream>>>(out_h, yW1, yb1, nullptr, nA, NODES);
    k_node_gemm<false><<<gN, blk, 0, stream>>>(nA,    yW2, yb2, nullptr, nB, NODES);
    k_node_gemm<false><<<gN, blk, 0, stream>>>(nB,    yW3, yb3, nullptr, hn, NODES);
    k_ydot<<<dim3((NODES + 255) / 256), dim3(256), 0, stream>>>(hn, yoW, yob, out_y);
}
